// Decision_Node_34832184770666
// MI455X (gfx1250) — compile-verified, matrix operand order checked by reading, not run on hardware
//
#include <hip/hip_runtime.h>

// CDNA5 / gfx1250: wave32, WMMA 16x16x32 bf16 with f32 accumulate.
// Block = 512 threads = 16 waves laid out 8(M) x 2(N).
// Each block computes a 128(M) x 256(N) tile of sp/gini; each wave owns
// 16 rows x 128 cols = 8 WMMA tiles (64 accumulator VGPRs -> high occupancy).
typedef __attribute__((ext_vector_type(16))) __bf16 v16bf;
typedef __attribute__((ext_vector_type(8)))  float  v8f;

#define BT_ROWS 131072   // B*T
#define TSZ     128
#define LSZ     256
#define FSZ     784
#define KC      32       // K-chunk per WMMA (bf16)
#define XSTR    40       // padded bf16 row stride in LDS (32 + 8) -> 80B, conflict-free
#define WSTR    40
#define NTILES  8        // N-tiles per wave (8 x 16 = 128 columns)

__device__ __forceinline__ unsigned short f2bf(float f) {
  union { float f; unsigned u; } v; v.f = f;
  unsigned r = v.u + 0x7FFFu + ((v.u >> 16) & 1u);   // round-to-nearest-even
  return (unsigned short)(r >> 16);
}

union BFrag { v16bf v; uint4 q[2]; };

__global__ __launch_bounds__(512)
void decision_node_fused(const float* __restrict__ x,
                         const float* __restrict__ W,
                         const float* __restrict__ bias,
                         const float* __restrict__ contrib,
                         float* __restrict__ out_sp,
                         float* __restrict__ out_gini)
{
  __shared__ unsigned short Xb[128 * XSTR];  // 10240 B
  __shared__ unsigned short Wb[256 * WSTR];  // 20480 B

  const int tid   = threadIdx.x;
  const int lane  = tid & 31;
  const int wave  = tid >> 5;                    // 0..15
  const int m0    = (wave & 7) * 16;             // wave's 16 rows of the 128-row tile
  const int n0    = (wave >> 3) * 128;           // wave's 128-column half
  const int l16   = lane & 15;
  const int lhalf = lane >> 4;                   // 0 or 1
  const long long row0 = (long long)blockIdx.x * 128;

  v8f acc[NTILES] = {};                          // 8 N-tiles of 16x16 f32

  for (int kc = 0; kc < FSZ; kc += KC) {
    __syncthreads();

    // ---- stage x tile: 128 x 32 f32 -> bf16 LDS (zero-pad K tail) ----
    #pragma unroll
    for (int i = 0; i < 2; ++i) {
      const int idx = tid + i * 512;             // 0..1023
      const int r   = idx >> 3;                  // 0..127
      const int q   = idx & 7;                   // float4 slot 0..7
      const int k   = kc + q * 4;
      ushort4 pk;
      if (k < FSZ) {
        const float4 v = *(const float4*)(x + (row0 + r) * FSZ + k);
        pk.x = f2bf(v.x); pk.y = f2bf(v.y); pk.z = f2bf(v.z); pk.w = f2bf(v.w);
      } else {
        pk.x = 0; pk.y = 0; pk.z = 0; pk.w = 0;
      }
      *(ushort4*)&Xb[r * XSTR + q * 4] = pk;
    }
    // ---- stage W tile: 256 x 32 f32 -> bf16 LDS ----
    #pragma unroll
    for (int i = 0; i < 4; ++i) {
      const int idx = tid + i * 512;             // 0..2047
      const int r   = idx >> 3;                  // 0..255
      const int q   = idx & 7;
      const int k   = kc + q * 4;
      ushort4 pk;
      if (k < FSZ) {
        const float4 v = *(const float4*)(W + r * FSZ + k);
        pk.x = f2bf(v.x); pk.y = f2bf(v.y); pk.z = f2bf(v.z); pk.w = f2bf(v.w);
      } else {
        pk.x = 0; pk.y = 0; pk.z = 0; pk.w = 0;
      }
      *(ushort4*)&Wb[r * WSTR + q * 4] = pk;
    }
    __syncthreads();

    // pull the next x chunk toward L2 while the math runs
    if (kc + KC < FSZ) {
      __builtin_prefetch(x + (row0 + (tid >> 2)) * FSZ + (kc + KC) + (tid & 3) * 8, 0, 1);
    }

    // ---- A fragment: 16x32 bf16 (lanes 0-15: K{0..7,16..23}; 16-31: K{8..15,24..31}) ----
    const int k0 = lhalf * 8;                    // bf16 element offset
    union { v16bf v; uint4 q[2]; } afrag;
    afrag.q[0] = *(const uint4*)&Xb[(m0 + l16) * XSTR + k0];
    afrag.q[1] = *(const uint4*)&Xb[(m0 + l16) * XSTR + k0 + 16];

    // ---- 8 N-tiles with software-pipelined (double-buffered) B fragments ----
    const unsigned short* wb_base = &Wb[(n0 + l16) * WSTR + k0];
    BFrag bfr[2];
    bfr[0].q[0] = *(const uint4*)(wb_base);
    bfr[0].q[1] = *(const uint4*)(wb_base + 16);
    #pragma unroll
    for (int nt = 0; nt < NTILES; ++nt) {
      const int cur = nt & 1;
      if (nt < NTILES - 1) {
        const unsigned short* p = wb_base + (nt + 1) * 16 * WSTR;
        bfr[cur ^ 1].q[0] = *(const uint4*)(p);
        bfr[cur ^ 1].q[1] = *(const uint4*)(p + 16);
      }
      acc[nt] = __builtin_amdgcn_wmma_f32_16x16x32_bf16(
          false, afrag.v, false, bfr[cur].v, (short)0, acc[nt], false, false);
    }
  }

  // ---- fused epilogue: bias + hardtanh + C=2 softmax Gini ----
  // C/D layout: lane half selects M offset 8, VGPR v = M row; N = n0 + nt*16 + (lane&15).
  #pragma unroll
  for (int nt = 0; nt < NTILES; ++nt) {
    const int N  = n0 + nt * 16 + l16;
    const float bN = bias[N];
    #pragma unroll
    for (int v = 0; v < 8; ++v) {
      const int M = v + 8 * lhalf;               // 0..15
      const int t = m0 + M;                      // block covers 128 rows -> t in [0,128)
      float s = acc[nt][v] + bN;
      s = fminf(1.0f, fmaxf(-1.0f, s));
      const float2 c01 = *(const float2*)(contrib + (size_t)(t * LSZ + N) * 2);
      const float d  = s * (c01.x - c01.y);      // softmax(C=2) == sigmoid of the gap
      const float p0 = 1.0f / (1.0f + __expf(-d));
      const float g  = 1.0f + 2.0f * p0 * (1.0f - p0);
      const size_t o = (size_t)(row0 + t) * LSZ + N;
      out_sp[o]   = s;
      out_gini[o] = g;
    }
  }
}

extern "C" void kernel_launch(void* const* d_in, const int* in_sizes, int n_in,
                              void* d_out, int out_size, void* d_ws, size_t ws_size,
                              hipStream_t stream) {
  const float* x       = (const float*)d_in[0];   // [B,T,F]
  const float* W       = (const float*)d_in[1];   // [L,F]
  const float* b       = (const float*)d_in[2];   // [L]
  const float* contrib = (const float*)d_in[3];   // [T,L,C]
  float* out_sp   = (float*)d_out;                           // [B,T,L]
  float* out_gini = (float*)d_out + (size_t)BT_ROWS * LSZ;   // [B,T,L]

  dim3 grid(BT_ROWS / 128);
  dim3 block(512);
  decision_node_fused<<<grid, block, 0, stream>>>(x, W, b, contrib, out_sp, out_gini);
}
